// MLPModel7_80109730005571
// MI455X (gfx1250) — compile-verified, matrix operand order checked by reading, not run on hardware
//
#include <hip/hip_runtime.h>
#include <hip/hip_bf16.h>

// ---------------------------------------------------------------------------
// MI455X / gfx1250 fused implementation.
//   dominant op: [2048,32768]x[32768,512] GEMM -> bf16 WMMA (16x16x32)
//   attention projection [512,64]x[64,64] per row -> bf16 WMMA too
//   big GEMM: double-buffered pipeline, A staged via
//   GLOBAL_LOAD_ASYNC_TO_LDS_B128 (ASYNCcnt), B transposed via registers
// ---------------------------------------------------------------------------

typedef __bf16 bf16;
typedef __attribute__((ext_vector_type(16))) __bf16 v16bf;
typedef __attribute__((ext_vector_type(8)))  __bf16 v8bf;
typedef __attribute__((ext_vector_type(8)))  float  v8f;

#define LN_EPS   1e-5f
#define B_TOT    2048
#define D_DIM    64
#define H_DIM    512
#define K_TOT    (D_DIM * H_DIM)   // 32768
#define CHUNK    1024              // batch rows per chunk (2 chunks)

// -------------------------------------------------------------------- utils
__device__ __forceinline__ float wave16_sum(float v) {
    // reduce across the 16 lanes of a half-wave (lanes sharing one C/D row)
    v += __shfl_xor(v, 1, 32);
    v += __shfl_xor(v, 2, 32);
    v += __shfl_xor(v, 4, 32);
    v += __shfl_xor(v, 8, 32);
    return v;
}

__device__ __forceinline__ float fast_tanh(float v) {
#if __has_builtin(__builtin_amdgcn_tanhf)
    return __builtin_amdgcn_tanhf(v);          // v_tanh_f32 (gfx1250 TRANS)
#elif __has_builtin(__builtin_amdgcn_tanh_f32)
    return __builtin_amdgcn_tanh_f32(v);
#else
    return tanhf(v);
#endif
}

// async global->LDS copy of 16B per lane (ASYNCcnt-tracked, bypasses VGPRs)
__device__ __forceinline__ void async_copy_b128(const void* gsrc, void* lds_dst) {
    unsigned long long ga = (unsigned long long)gsrc;   // 64-bit vaddr pair
    unsigned int       la = (unsigned int)(unsigned long long)lds_dst;
    asm volatile("global_load_async_to_lds_b128 %0, %1, off"
                 :: "v"(la), "v"(ga) : "memory");
}
__device__ __forceinline__ void async_wait0() {
    asm volatile("s_wait_asynccnt 0" ::: "memory");
}

// ---------------------------------------------------------------- kernel Z
__global__ void k_init_out(float* __restrict__ out, const float* __restrict__ bo, int n) {
    int i = blockIdx.x * blockDim.x + threadIdx.x;
    if (i < n) out[i] = bo[0];
}

// ---------------------------------------------------------------- kernel A
// fp32 -> bf16 conversion of Wh (done once; Wh_bf16 = 32 MB, L2-resident)
__global__ void k_convert_wh(const float* __restrict__ Wh, bf16* __restrict__ WhB, int n) {
    int i = (blockIdx.x * blockDim.x + threadIdx.x) * 8;
    if (i + 8 > n) return;
    float4 f0 = *(const float4*)(Wh + i);
    float4 f1 = *(const float4*)(Wh + i + 4);
    v8bf o;
    o[0] = (bf16)f0.x; o[1] = (bf16)f0.y; o[2] = (bf16)f0.z; o[3] = (bf16)f0.w;
    o[4] = (bf16)f1.x; o[5] = (bf16)f1.y; o[6] = (bf16)f1.z; o[7] = (bf16)f1.w;
    *(v8bf*)(WhB + i) = o;
}

// ---------------------------------------------------------------- kernel B
// One workgroup per batch row:
//   xp = relu(x[d]*W_in[d,h]+b_in[d,h])           -> LDS (bf16, [512][64])
//   u  = tanh(xp @ Wf + bWf)                       -> WMMA bf16
//   scores[h] = u . Vf  (bVf is softmax-invariant) -> reg partials + ds_add_f32
//   attn = softmax_h(scores)
//   xx[b, h*64+d] = LayerNorm_d( xp*(1+attn[h]) ) * gamma + beta  (bf16 out)
__global__ void __launch_bounds__(256)
k_rowfused(const float* __restrict__ x,    const float* __restrict__ W_in,
           const float* __restrict__ b_in, const float* __restrict__ Wf,
           const float* __restrict__ bWf,  const float* __restrict__ Vf,
           const float* __restrict__ gamma,const float* __restrict__ beta,
           bf16* __restrict__ xx, int row_base)
{
    __shared__ bf16  s_xp[H_DIM][D_DIM];   // 64 KB, pitch 128B (16B aligned)
    __shared__ bf16  s_wft[D_DIM][80];     // Wf transposed: s_wft[n][k], 10 KB
    __shared__ float s_x[D_DIM];
    __shared__ float s_sc[H_DIM];
    __shared__ float s_red[8];
    __shared__ float s_g[D_DIM], s_b[D_DIM];

    const int tid  = threadIdx.x;
    const int b    = row_base + blockIdx.x;
    const int wave = tid >> 5;
    const int lane = tid & 31;
    const int lrow = lane & 15;
    const int lhi  = lane >> 4;

    if (tid < D_DIM) {
        s_x[tid] = x[(size_t)b * D_DIM + tid];
        s_g[tid] = gamma[tid];
        s_b[tid] = beta[tid];
    }
    for (int i = tid; i < D_DIM * D_DIM; i += 256) {
        int d = i >> 6, e = i & 63;            // Wf[d][e]
        s_wft[e][d] = (bf16)Wf[i];             // store transposed for B frags
    }
    for (int i = tid; i < H_DIM; i += 256) s_sc[i] = 0.f;

    // hoist Vf/bWf for this lane's 4 possible N columns
    float vfn[4], bwn[4];
    #pragma unroll
    for (int nt = 0; nt < 4; ++nt) {
        vfn[nt] = Vf[nt * 16 + lrow];
        bwn[nt] = bWf[nt * 16 + lrow];
    }
    __syncthreads();

    // xp = relu(x*W_in + b_in); W_in/b_in are [D][H] row-major -> coalesced
    for (int i = tid; i < D_DIM * H_DIM; i += 256) {
        int d = i >> 9;          // / 512
        int h = i & 511;
        float v = fmaf(s_x[d], W_in[i], b_in[i]);
        s_xp[h][d] = (bf16)fmaxf(v, 0.f);
    }
    __syncthreads();

    // ---- WMMA: u = xp @ Wf, then tanh + Vf-dot into s_sc ------------------
    // 32 M-tiles (h) x 4 N-tiles (e), K=64 = 2 wmma steps. 8 waves x 4 mtiles.
    for (int mt = 0; mt < 4; ++mt) {
        const int m0 = (wave * 4 + mt) * 16;
        const bf16* ap = &s_xp[m0 + lrow][0];
        v16bf a0, a1;   // A frags for k=[0,32) and [32,64) per ISA 16-bit layout
        ((uint4*)&a0)[0] = *(const uint4*)(ap + lhi * 8);
        ((uint4*)&a0)[1] = *(const uint4*)(ap + 16 + lhi * 8);
        ((uint4*)&a1)[0] = *(const uint4*)(ap + 32 + lhi * 8);
        ((uint4*)&a1)[1] = *(const uint4*)(ap + 48 + lhi * 8);
        float cr[8];
        #pragma unroll
        for (int r = 0; r < 8; ++r) cr[r] = 0.f;
        #pragma unroll
        for (int nt = 0; nt < 4; ++nt) {
            const bf16* bp = &s_wft[nt * 16 + lrow][0];
            v16bf b0, b1;
            ((uint4*)&b0)[0] = *(const uint4*)(bp + lhi * 8);
            ((uint4*)&b0)[1] = *(const uint4*)(bp + 16 + lhi * 8);
            ((uint4*)&b1)[0] = *(const uint4*)(bp + 32 + lhi * 8);
            ((uint4*)&b1)[1] = *(const uint4*)(bp + 48 + lhi * 8);
            v8f acc = {0.f, 0.f, 0.f, 0.f, 0.f, 0.f, 0.f, 0.f};
            acc = __builtin_amdgcn_wmma_f32_16x16x32_bf16(false, a0, false, b0,
                                                          (short)0, acc, false, false);
            acc = __builtin_amdgcn_wmma_f32_16x16x32_bf16(false, a1, false, b1,
                                                          (short)0, acc, false, false);
            #pragma unroll
            for (int r = 0; r < 8; ++r)          // D row m = m0 + r + 8*lhi
                cr[r] += fast_tanh(acc[r] + bwn[nt]) * vfn[nt];
        }
        // 16 lanes share each row -> LDS float atomic accumulate (no shuffles)
        #pragma unroll
        for (int r = 0; r < 8; ++r)
            atomicAdd(&s_sc[m0 + r + 8 * lhi], cr[r]);
    }
    __syncthreads();

    // ---- softmax over H=512 (bVf constant -> cancels) ---------------------
    float mloc = -1e30f;
    for (int i = tid; i < H_DIM; i += 256) mloc = fmaxf(mloc, s_sc[i]);
    for (int off = 16; off; off >>= 1) mloc = fmaxf(mloc, __shfl_xor(mloc, off, 32));
    if (lane == 0) s_red[wave] = mloc;
    __syncthreads();
    float mx = s_red[0];
    #pragma unroll
    for (int w = 1; w < 8; ++w) mx = fmaxf(mx, s_red[w]);

    float sloc = 0.f;
    for (int i = tid; i < H_DIM; i += 256) {
        float e = __expf(s_sc[i] - mx);
        s_sc[i] = e;
        sloc += e;
    }
    for (int off = 16; off; off >>= 1) sloc += __shfl_xor(sloc, off, 32);
    __syncthreads();
    if (lane == 0) s_red[wave] = sloc;
    __syncthreads();
    float ssum = 0.f;
    #pragma unroll
    for (int w = 0; w < 8; ++w) ssum += s_red[w];
    const float inv = 1.0f / ssum;

    // ---- (1+attn) scale + LayerNorm over D, emit bf16 xx ------------------
    for (int h = tid; h < H_DIM; h += 256) {
        const float a = 1.0f + s_sc[h] * inv;
        float sum = 0.f, sq = 0.f;
        for (int d = 0; d < D_DIM; ++d) {
            float v = (float)s_xp[h][d] * a;
            sum += v; sq += v * v;
        }
        const float mu  = sum * (1.0f / D_DIM);
        const float var = sq * (1.0f / D_DIM) - mu * mu;
        const float rs  = rsqrtf(var + LN_EPS);
        bf16* dst = xx + (size_t)blockIdx.x * K_TOT + (size_t)h * D_DIM;
        for (int d = 0; d < D_DIM; ++d) {
            float v = (float)s_xp[h][d] * a;
            dst[d] = (bf16)(((v - mu) * rs) * s_g[d] + s_b[d]);
        }
    }
}

// ---------------------------------------------------------------- kernel C
// hid = relu(xx @ Wh + bh); out[row] += hid . Wo   (global f32 atomics)
// WG tile 128x128, 8 waves (2x4), each wave 64x32 -> 8 WMMAs per k-step.
// Double-buffered: A tile k+1 via GLOBAL_LOAD_ASYNC_TO_LDS_B128 and B tile
// k+1 via registers, both issued before computing tile k.
__global__ void __launch_bounds__(256)
k_gemm(const bf16* __restrict__ xx, const bf16* __restrict__ WhB,
       const float* __restrict__ bh, const float* __restrict__ Wo,
       float* __restrict__ out, int row_base)
{
    __shared__ bf16 sA [2][128][40];   // [m][k], pitch 80B (16B aligned)
    __shared__ bf16 sBt[2][128][40];   // [n][k], transposed Wh tile

    const int tid  = threadIdx.x;
    const int wave = tid >> 5;
    const int lane = tid & 31;
    const int lrow = lane & 15;
    const int lhi  = lane >> 4;
    const int wrow = wave >> 2;            // 0..1
    const int wcol = wave & 3;             // 0..3
    const int m_wg = blockIdx.x * 128;     // chunk-local row base
    const int n_wg = blockIdx.y * 128;

    v8f acc[4][2] = {};

    const int a_r  = tid >> 1;             // A stage: 128 rows x 32 k
    const int a_ko = (tid & 1) * 16;
    const int b_k  = tid >> 3;             // B stage: 32 k-rows x 128 n
    const int b_no = (tid & 7) * 16;

    const bf16* srcA0 = xx  + (size_t)(m_wg + a_r) * K_TOT + a_ko;
    const bf16* srcB0 = WhB + (size_t)b_k * H_DIM + n_wg + b_no;

    union BStage { uint4 q[2]; bf16 e[16]; };

    // ---- prologue: stage tile 0 into buffer 0 -----------------------------
    {
        async_copy_b128(srcA0,     &sA[0][a_r][a_ko]);
        async_copy_b128(srcA0 + 8, &sA[0][a_r][a_ko + 8]);
        BStage s;
        s.q[0] = *(const uint4*)srcB0;
        s.q[1] = *(const uint4*)(srcB0 + 8);
        #pragma unroll
        for (int j = 0; j < 16; ++j) sBt[0][b_no + j][b_k] = s.e[j];
    }
    async_wait0();
    __syncthreads();

    const int NK = K_TOT / 32;
    for (int kt = 0; kt < NK; ++kt) {
        const int cur = kt & 1, nxt = cur ^ 1;
        const bool have_next = (kt + 1 < NK);

        // issue next tile's loads before computing (latency hiding)
        BStage s;
        if (have_next) {
            const int k1 = (kt + 1) * 32;
            async_copy_b128(srcA0 + k1,     &sA[nxt][a_r][a_ko]);
            async_copy_b128(srcA0 + k1 + 8, &sA[nxt][a_r][a_ko + 8]);
            const bf16* srcB = srcB0 + (size_t)k1 * H_DIM;
            s.q[0] = *(const uint4*)srcB;
            s.q[1] = *(const uint4*)(srcB + 8);
            if (kt + 2 < NK)            // exercise global_prefetch_b8
                __builtin_prefetch((const void*)(srcB + (size_t)32 * H_DIM), 0, 0);
        }

        // compute tile kt from buffer `cur`
        v16bf af[4];
        #pragma unroll
        for (int mt = 0; mt < 4; ++mt) {
            const bf16* ap = &sA[cur][wrow * 64 + mt * 16 + lrow][0];
            ((uint4*)&af[mt])[0] = *(const uint4*)(ap + lhi * 8);
            ((uint4*)&af[mt])[1] = *(const uint4*)(ap + 16 + lhi * 8);
        }
        #pragma unroll
        for (int nt = 0; nt < 2; ++nt) {
            const bf16* bp = &sBt[cur][wcol * 32 + nt * 16 + lrow][0];
            v16bf bfr;
            ((uint4*)&bfr)[0] = *(const uint4*)(bp + lhi * 8);
            ((uint4*)&bfr)[1] = *(const uint4*)(bp + 16 + lhi * 8);
            #pragma unroll
            for (int mt = 0; mt < 4; ++mt)
                acc[mt][nt] = __builtin_amdgcn_wmma_f32_16x16x32_bf16(
                    false, af[mt], false, bfr, (short)0, acc[mt][nt], false, false);
        }

        // scatter next transposed B tile, then make buffers visible
        if (have_next) {
            #pragma unroll
            for (int j = 0; j < 16; ++j) sBt[nxt][b_no + j][b_k] = s.e[j];
        }
        async_wait0();
        __syncthreads();
    }

    // epilogue: relu(+bh), fold @Wo, atomic accumulate per output row
    #pragma unroll
    for (int mt = 0; mt < 4; ++mt) {
        #pragma unroll
        for (int nt = 0; nt < 2; ++nt) {
            const int n = n_wg + wcol * 32 + nt * 16 + lrow;
            const float bhv = bh[n];
            const float wov = Wo[n];
            v8f a = acc[mt][nt];
            #pragma unroll
            for (int r = 0; r < 8; ++r) {
                const int m = m_wg + wrow * 64 + mt * 16 + r + 8 * lhi;
                float hid = fmaxf(a[r] + bhv, 0.f);
                float c = wave16_sum(hid * wov);
                if (lrow == 0) atomicAdd(&out[row_base + m], c);
            }
        }
    }
}

// ------------------------------------------------------------------ launch
extern "C" void kernel_launch(void* const* d_in, const int* in_sizes, int n_in,
                              void* d_out, int out_size, void* d_ws, size_t ws_size,
                              hipStream_t stream) {
    (void)in_sizes; (void)n_in; (void)out_size; (void)ws_size;
    const float* x     = (const float*)d_in[0];
    const float* W_in  = (const float*)d_in[1];
    const float* b_in  = (const float*)d_in[2];
    const float* Wf    = (const float*)d_in[3];
    const float* bWf   = (const float*)d_in[4];
    const float* Vf    = (const float*)d_in[5];
    // d_in[6] = bVf: constant over the softmax axis -> cancels, unused
    const float* gamma = (const float*)d_in[7];
    const float* beta  = (const float*)d_in[8];
    const float* Wh    = (const float*)d_in[9];
    const float* bh    = (const float*)d_in[10];
    const float* Wo    = (const float*)d_in[11];
    const float* bo    = (const float*)d_in[12];
    float* out = (float*)d_out;

    bf16* WhB = (bf16*)d_ws;                                       // 32 MB
    bf16* xxB = (bf16*)((char*)d_ws + (size_t)K_TOT * H_DIM * 2);  // 64 MB

    k_init_out<<<(B_TOT + 255) / 256, 256, 0, stream>>>(out, bo, B_TOT);
    k_convert_wh<<<(K_TOT * H_DIM) / (256 * 8), 256, 0, stream>>>(Wh, WhB, K_TOT * H_DIM);

    for (int c = 0; c < B_TOT; c += CHUNK) {
        k_rowfused<<<CHUNK, 256, 0, stream>>>(x, W_in, b_in, Wf, bWf, Vf,
                                              gamma, beta, xxB, c);
        k_gemm<<<dim3(CHUNK / 128, H_DIM / 128), 256, 0, stream>>>(xxB, WhB, bh,
                                                                   Wo, out, c);
    }
}
